// Decoder_57320633532534
// MI455X (gfx1250) — matrix operand order repaired and checked by reading, not compiled, not run on hardware
//
#include <hip/hip_runtime.h>

// ---------------------------------------------------------------------------
// CDNA5 / gfx1250 implementation of the RGCN shot-type decoder.
// Math restructure:  sum_r A_r x W_r + x S^T  ==  sum_{j=0..6} (M_j x) Wcat_j
// with M_j = sum_r comb[r,j] A_r  (j<6),  M_6 = I,  Wcat = [basis_0..5 ; S^T].
// All heavy math runs through v_wmma_f32_16x16x32_f16 (f16 in, f32 acc).
// gemm_big stages A/B panels through LDS (ds_load_b128 fragments) so L2
// traffic per WMMA drops ~5x vs direct per-wave global fragment loads.
// ---------------------------------------------------------------------------

typedef _Float16 half8 __attribute__((ext_vector_type(8)));
typedef _Float16 v16h  __attribute__((ext_vector_type(16)));
typedef float    v8f   __attribute__((ext_vector_type(8)));

#define BATCH 256
#define HID   512
#define NODE  64      // padded node count (62 or 63 real)
#define KCAT  3584    // 7*512
#define NREL  12
#define NBAS  6
#define NTYPE 11

#define LDS_PAD   8                 // halves; row stride 40 -> conflict-free frags
#define AS_LD     (32 + LDS_PAD)    // 40
#define BS_LD     (32 + LDS_PAD)    // 40

// ---------------- WMMA fragment helpers (ISA 7.12.2 layouts) ----------------

// A fragment: 16x32 f16 tile from row-major matrix (ld halves per row).
// lane L: M = r0 + L%16 ; halves[0..7] = K = k0+d .. ; halves[8..15] = K = k0+16+d
__device__ __forceinline__ v16h ld_frag_a(const _Float16* base, int ld, int r0, int k0) {
    const int lane = threadIdx.x & 31;
    const int m = r0 + (lane & 15);
    const int d = (lane >> 4) * 8;
    const half8 x = *reinterpret_cast<const half8*>(base + (size_t)m * ld + k0 + d);
    const half8 y = *reinterpret_cast<const half8*>(base + (size_t)m * ld + k0 + 16 + d);
    v16h r;
#pragma unroll
    for (int i = 0; i < 8; ++i) { r[i] = x[i]; r[i + 8] = y[i]; }
    return r;
}

// B fragment: 32x16 f16 tile; B stored TRANSPOSED (ld halves per N-row).
// lane L: N = n0 + L%16 ; halves[0..7] = K = k0+g+d ; halves[8..15] = K = k0+16+g+d  (g = 8 for lanes >=16)
__device__ __forceinline__ v16h ld_frag_b(const _Float16* baseT, int ld, int n0, int k0) {
    const int lane = threadIdx.x & 31;
    const int n = n0 + (lane & 15);
    const int g = (lane >> 4) * 8;
    const half8 x = *reinterpret_cast<const half8*>(baseT + (size_t)n * ld + k0 + g);
    const half8 y = *reinterpret_cast<const half8*>(baseT + (size_t)n * ld + k0 + 16 + g);
    v16h r;
#pragma unroll
    for (int i = 0; i < 8; ++i) { r[i] = x[i]; r[i + 8] = y[i]; }
    return r;
}

__device__ __forceinline__ v8f wmma_f16(v16h a, v16h b, v8f c) {
    return __builtin_amdgcn_wmma_f32_16x16x32_f16(false, a, false, b, (short)0, c, false, false);
}

__device__ __forceinline__ float apply_act(float v, int act) {
    if (act == 1) return v > 0.0f ? v : 0.0f;
    if (act == 2) return 1.0f / (1.0f + __expf(-v));
    return v;
}

__device__ __forceinline__ v8f zero8() {
    v8f z;
#pragma unroll
    for (int i = 0; i < 8; ++i) z[i] = 0.0f;
    return z;
}

// ---------------- tiny front-end: initial node embedding ----------------
// initialH[b][i][h] (f16), i=0:playerA row, i=1:playerB row
__global__ __launch_bounds__(256) void initial_kernel(
    const int* player, const float* Ax, const float* Ay, const float* Bx, const float* By,
    const float* emb_table, const float* coord_W, const float* coord_b,
    const float* in_W, const float* in_b, _Float16* initialH)
{
    const int b = blockIdx.x;
    const int i = blockIdx.y;
    __shared__ float feat[64];
    const int t = threadIdx.x;
    if (t < 32) {
        const float x = (i == 0) ? Ax[b] : Bx[b];
        const float y = (i == 0) ? Ay[b] : By[b];
        float v = coord_W[t * 2 + 0] * x + coord_W[t * 2 + 1] * y + coord_b[t];
        feat[t] = v > 0.0f ? v : 0.0f;
    } else if (t < 64) {
        const int p = player[b * 2 + i];
        feat[t] = emb_table[p * 32 + (t - 32)];
    }
    __syncthreads();
#pragma unroll
    for (int rep = 0; rep < 2; ++rep) {
        const int h = t + rep * 256;
        float acc = in_b[h];
#pragma unroll
        for (int f = 0; f < 64; ++f) acc += feat[f] * in_W[h * 64 + f];
        initialH[((size_t)b * 2 + i) * HID + h] = (_Float16)acc;
    }
}

// ---------------- pack X (transposed [b][hid][node], f16) ----------------
__global__ __launch_bounds__(256) void pack_x1_kernel(
    const float* enc, const _Float16* initialH, _Float16* XT)
{
    const size_t idx = (size_t)blockIdx.x * 256 + threadIdx.x;  // B*512*64
    const int k = idx & 63;
    const int h = (idx >> 6) & 511;
    const int b = idx >> 15;
    float v = 0.0f;
    if (k < 60)      v = enc[((size_t)b * 60 + k) * HID + h];
    else if (k < 62) v = (float)initialH[((size_t)b * 2 + (k - 60)) * HID + h];
    XT[idx] = (_Float16)v;
}

__global__ __launch_bounds__(256) void pack_x2_kernel(
    const float* enc, const _Float16* T1T, const _Float16* initialH, _Float16* XT)
{
    const size_t idx = (size_t)blockIdx.x * 256 + threadIdx.x;
    const int k = idx & 63;
    const int h = (idx >> 6) & 511;
    const int b = idx >> 15;
    _Float16 v = (_Float16)0.0f;
    if (k < 58)      v = (_Float16)enc[((size_t)b * 60 + k) * HID + h];
    else if (k < 62) v = T1T[((size_t)b * HID + h) * NODE + k];        // rgcn1 rows 58..61
    else if (k == 62) v = initialH[((size_t)b * 2 + 1) * HID + h];     // initial[:,1]
    XT[idx] = v;
}

// ---------------- weight repack: WcatT[n][k] f16 (k = j*512+h; j==6 -> self^T)
__global__ __launch_bounds__(256) void build_wcat_kernel(
    const float* basis, const float* selfm, _Float16* WcatT)
{
    const size_t idx = (size_t)blockIdx.x * 256 + threadIdx.x;  // 512*3584
    const int k = idx % KCAT;
    const int n = idx / KCAT;
    const int j = k >> 9;
    const int h = k & 511;
    float v;
    if (j < NBAS) v = basis[((size_t)j * HID + h) * HID + n];
    else          v = selfm[(size_t)n * HID + h];   // (x @ self^T)[k] uses self[k][h]
    WcatT[(size_t)n * KCAT + k] = (_Float16)v;
}

// ---------------- M matrices for rgcn #1: Mbuf[b][layer][j][64][64] f16 -----
__global__ __launch_bounds__(256) void build_m1_kernel(
    const int* adj, const float* comb0, const float* comb1, _Float16* Mbuf)
{
    const int b = blockIdx.x;
    __shared__ float c0[NREL * NBAS], c1[NREL * NBAS];
    if (threadIdx.x < NREL * NBAS) c0[threadIdx.x] = comb0[threadIdx.x];
    else if (threadIdx.x < 2 * NREL * NBAS) c1[threadIdx.x - NREL * NBAS] = comb1[threadIdx.x - NREL * NBAS];
    __syncthreads();
    for (int e = threadIdx.x; e < NODE * NODE; e += 256) {
        const int n = e >> 6, m = e & 63;
        float a[NREL];
        const bool valid = (n < 62) && (m < 62);
#pragma unroll
        for (int r = 0; r < NREL; ++r)
            a[r] = valid ? (float)adj[((size_t)b * 13 + 1 + r) * 3844 + n * 62 + m] : 0.0f;
        for (int l = 0; l < 2; ++l) {
            const float* cc = (l == 0) ? c0 : c1;
            _Float16* M = Mbuf + (((size_t)b * 2 + l) * 7) * 4096;
#pragma unroll
            for (int j = 0; j < NBAS; ++j) {
                float s = 0.0f;
#pragma unroll
                for (int r = 0; r < NREL; ++r) s += cc[r * NBAS + j] * a[r];
                M[(size_t)j * 4096 + e] = (_Float16)s;
            }
            M[(size_t)6 * 4096 + e] = (_Float16)((n == m) ? 1.0f : 0.0f);  // identity slot
        }
    }
}

// ---------------- M matrices for rgcn #2 (update_adjacency + drop idx 62) ---
__device__ __forceinline__ bool mask11(int n, int m) {
    const int d = n > m ? n - m : m - n;
    const int mn = n < m ? n : m;
    return d == 2 && ((mn & 3) == 0 || (mn & 3) == 3);
}
__device__ __forceinline__ bool mask12(int n, int m) {
    const int d = n > m ? n - m : m - n;
    const int mn = n < m ? n : m;
    return d == 2 && ((mn & 3) == 1 || (mn & 3) == 2);
}

__global__ __launch_bounds__(256) void build_m2_kernel(
    const int* adj, const float* comb0, const float* comb1, _Float16* Mbuf)
{
    const int b = blockIdx.x;
    __shared__ float c0[NREL * NBAS], c1[NREL * NBAS];
    if (threadIdx.x < NREL * NBAS) c0[threadIdx.x] = comb0[threadIdx.x];
    else if (threadIdx.x < 2 * NREL * NBAS) c1[threadIdx.x - NREL * NBAS] = comb1[threadIdx.x - NREL * NBAS];
    __syncthreads();
    for (int e = threadIdx.x; e < NODE * NODE; e += 256) {
        const int np = e >> 6, mp = e & 63;        // indices in the 63-node graph (63 = pad)
        float a[NREL];
        if (np >= 63 || mp >= 63) {
#pragma unroll
            for (int r = 0; r < NREL; ++r) a[r] = 0.0f;
        } else {
            const int nn = (np == 62) ? 63 : np;   // keep = [0..61, 63]
            const int mm = (mp == 62) ? 63 : mp;
#pragma unroll
            for (int r = 0; r < NREL; ++r) {
                const int c = r + 1;
                float v = (nn < 62 && mm < 62)
                              ? (float)adj[((size_t)b * 13 + c) * 3844 + nn * 62 + mm]
                              : 0.0f;
                if (c == 11 && mask11(nn, mm)) v = 1.0f;   // new[:,11] |= m11
                if (c == 12 && mask12(nn, mm)) v = 1.0f;   // new[:,12] |= m12
                a[r] = v;
            }
        }
        for (int l = 0; l < 2; ++l) {
            const float* cc = (l == 0) ? c0 : c1;
            _Float16* M = Mbuf + (((size_t)b * 2 + l) * 7) * 4096;
#pragma unroll
            for (int j = 0; j < NBAS; ++j) {
                float s = 0.0f;
#pragma unroll
                for (int r = 0; r < NREL; ++r) s += cc[r * NBAS + j] * a[r];
                M[(size_t)j * 4096 + e] = (_Float16)s;
            }
            M[(size_t)6 * 4096 + e] = (_Float16)((np == mp) ? 1.0f : 0.0f);
        }
    }
}

// ---------------- GEMM 1: Ycat[b, node, j*512+n] = M_j(64x64) @ X(64x512) ---
// one wave = one 16x16 tile, K=64 (2 wmma). waves = B*7*4*32
__global__ __launch_bounds__(256) void gemm_mx_kernel(
    const _Float16* Mbuf, int layer, const _Float16* XT, _Float16* Ycat)
{
    const int wave = blockIdx.x * 8 + (threadIdx.x >> 5);
    const int nt = wave & 31;
    const int rt = (wave >> 5) & 3;
    const int j  = (wave >> 7) % 7;
    const int b  = wave / (32 * 4 * 7);
    const _Float16* M = Mbuf + (((size_t)b * 2 + layer) * 7 + j) * 4096;
    const _Float16* X = XT + (size_t)b * HID * NODE;   // transposed: [hid][node]
    v8f c = zero8();
#pragma unroll
    for (int k0 = 0; k0 < 64; k0 += 32) {
        const v16h af = ld_frag_a(M, NODE, rt * 16, k0);
        const v16h bf = ld_frag_b(X, NODE, nt * 16, k0);
        c = wmma_f16(af, bf, c);
    }
    // row-major store into Ycat (rows = node, ld = KCAT)
    const int lane = threadIdx.x & 31;
    const int col = j * HID + nt * 16 + (lane & 15);
    const int rbase = rt * 16 + (lane >> 4) * 8;
    _Float16* Y = Ycat + (size_t)b * NODE * KCAT;
#pragma unroll
    for (int v = 0; v < 8; ++v)
        Y[(size_t)(rbase + v) * KCAT + col] = (_Float16)c[v];
}

// ---------------- GEMM 2: out = act( Ycat(B*64 x 3584) @ Wcat(3584 x 512) ) -
// One 256-thread block = one batch x 128 output cols (64x128 tile).
// Per k-step (K=32): stage A panel 64x32 + B panel 128x32 through LDS,
// 8 waves each compute a 32x32 sub-tile (4 wmma) from ds_load_b128 fragments.
// Global loads for step k+1 are issued before the compute of step k.
__global__ __launch_bounds__(256) void gemm_big_kernel(
    const _Float16* Ycat, const _Float16* WcatT,
    _Float16* outT, float* outTf, int act)
{
    __shared__ _Float16 As[64 * AS_LD];    // 5120 B
    __shared__ _Float16 Bs[128 * BS_LD];   // 10240 B
    const int b  = blockIdx.x >> 2;
    const int C0 = (blockIdx.x & 3) * 128;
    const int t = threadIdx.x;
    const int wave = t >> 5;
    const int lane = t & 31;
    const _Float16* A = Ycat + (size_t)b * NODE * KCAT;

    // staging assignment: thread t loads row t/4, k-chunk (t%4)*8
    const int srow = t >> 2;
    const int skc  = (t & 3) * 8;
    const _Float16* gA  = A + (size_t)srow * KCAT + skc;
    const _Float16* gB0 = WcatT + (size_t)(C0 + srow) * KCAT + skc;
    const _Float16* gB1 = WcatT + (size_t)(C0 + srow + 64) * KCAT + skc;

    const int R0  = (wave >> 2) * 32;   // wave row block (0 or 32)
    const int CC0 = (wave & 3) * 32;    // wave col block within the 128 cols

    v8f c00 = zero8(), c01 = zero8(), c10 = zero8(), c11 = zero8();

    half8 ra  = *reinterpret_cast<const half8*>(gA);
    half8 rb0 = *reinterpret_cast<const half8*>(gB0);
    half8 rb1 = *reinterpret_cast<const half8*>(gB1);

    for (int ks = 0; ks < KCAT / 32; ++ks) {
        __syncthreads();   // previous compute done; LDS writable
        *reinterpret_cast<half8*>(&As[srow * AS_LD + skc]) = ra;
        *reinterpret_cast<half8*>(&Bs[srow * BS_LD + skc]) = rb0;
        *reinterpret_cast<half8*>(&Bs[(srow + 64) * BS_LD + skc]) = rb1;
        if (ks + 1 < KCAT / 32) {   // prefetch next panel into registers
            const int k1 = (ks + 1) * 32;
            ra  = *reinterpret_cast<const half8*>(gA + k1);
            rb0 = *reinterpret_cast<const half8*>(gB0 + k1);
            rb1 = *reinterpret_cast<const half8*>(gB1 + k1);
        }
        __syncthreads();   // panel visible
        const v16h a0 = ld_frag_a(&As[0], AS_LD, R0, 0);
        const v16h a1 = ld_frag_a(&As[0], AS_LD, R0 + 16, 0);
        const v16h b0 = ld_frag_b(&Bs[0], BS_LD, CC0, 0);
        const v16h b1 = ld_frag_b(&Bs[0], BS_LD, CC0 + 16, 0);
        c00 = wmma_f16(a0, b0, c00);
        c01 = wmma_f16(a0, b1, c01);
        c10 = wmma_f16(a1, b0, c10);
        c11 = wmma_f16(a1, b1, c11);
    }

    const v8f cs[4] = {c00, c01, c10, c11};
#pragma unroll
    for (int mt = 0; mt < 2; ++mt) {
#pragma unroll
        for (int nt = 0; nt < 2; ++nt) {
            const v8f c = cs[mt * 2 + nt];
            const int h = C0 + CC0 + nt * 16 + (lane & 15);
            const int node0 = R0 + mt * 16 + (lane >> 4) * 8;
            float vals[8];
#pragma unroll
            for (int v = 0; v < 8; ++v) vals[v] = apply_act(c[v], act);
            if (outT) {  // transposed f16 store: one 16B store per lane
                half8 hh;
#pragma unroll
                for (int v = 0; v < 8; ++v) hh[v] = (_Float16)vals[v];
                *reinterpret_cast<half8*>(outT + ((size_t)b * HID + h) * NODE + node0) = hh;
            }
            if (outTf) {
                float4 f0 = make_float4(vals[0], vals[1], vals[2], vals[3]);
                float4 f1 = make_float4(vals[4], vals[5], vals[6], vals[7]);
                float* p = outTf + ((size_t)b * HID + h) * NODE + node0;
                *reinterpret_cast<float4*>(p) = f0;
                *reinterpret_cast<float4*>(p + 4) = f1;
            }
        }
    }
}

// ---------------- final logits: [black=row62, white=row60] @ type_W^T + b ---
__global__ void final_kernel(const float* T2T, const float* type_W,
                             const float* type_b, float* out)
{
    const int b = blockIdx.x;
    const int t = threadIdx.x;
    if (t >= NTYPE) return;
    float acc = type_b[t];
    const float* base = T2T + (size_t)b * HID * NODE;
    for (int h = 0; h < HID; ++h) {
        acc += base[(size_t)h * NODE + 62] * type_W[t * 1024 + h];        // black
        acc += base[(size_t)h * NODE + 60] * type_W[t * 1024 + 512 + h];  // white
    }
    out[b * NTYPE + t] = acc;
}

// ---------------------------------------------------------------------------
extern "C" void kernel_launch(void* const* d_in, const int* in_sizes, int n_in,
                              void* d_out, int out_size, void* d_ws, size_t ws_size,
                              hipStream_t stream) {
    (void)in_sizes; (void)n_in; (void)out_size; (void)ws_size;
    const int*   player   = (const int*)d_in[0];
    /* d_in[1] = step == 32 (compile-time constant of the harness) */
    const float* enc      = (const float*)d_in[2];
    const int*   adj      = (const int*)d_in[3];
    const float* Ax       = (const float*)d_in[4];
    const float* Ay       = (const float*)d_in[5];
    const float* Bx       = (const float*)d_in[6];
    const float* By       = (const float*)d_in[7];
    const float* emb      = (const float*)d_in[8];
    const float* coord_W  = (const float*)d_in[9];
    const float* coord_b  = (const float*)d_in[10];
    const float* in_W     = (const float*)d_in[11];
    const float* in_b     = (const float*)d_in[12];
    const float* basis0   = (const float*)d_in[13];
    const float* comb0    = (const float*)d_in[14];
    const float* self0    = (const float*)d_in[15];
    const float* basis1   = (const float*)d_in[16];
    const float* comb1    = (const float*)d_in[17];
    const float* self1    = (const float*)d_in[18];
    const float* type_W   = (const float*)d_in[19];
    const float* type_b   = (const float*)d_in[20];
    float* out = (float*)d_out;

    // ---- carve workspace ----
    char* ws = (char*)d_ws;
    size_t off = 0;
    auto carve = [&](size_t bytes) { char* p = ws + off; off += (bytes + 255) & ~(size_t)255; return p; };
    _Float16* initialH = (_Float16*)carve((size_t)BATCH * 2 * HID * 2);
    _Float16* XT1      = (_Float16*)carve((size_t)BATCH * HID * NODE * 2);
    _Float16* XT2      = (_Float16*)carve((size_t)BATCH * HID * NODE * 2);
    _Float16* M1       = (_Float16*)carve((size_t)BATCH * 2 * 7 * 4096 * 2);
    _Float16* M2       = (_Float16*)carve((size_t)BATCH * 2 * 7 * 4096 * 2);
    _Float16* W0T      = (_Float16*)carve((size_t)HID * KCAT * 2);
    _Float16* W1T      = (_Float16*)carve((size_t)HID * KCAT * 2);
    _Float16* Ycat     = (_Float16*)carve((size_t)BATCH * NODE * KCAT * 2);
    _Float16* HT       = (_Float16*)carve((size_t)BATCH * HID * NODE * 2);
    _Float16* T1T      = (_Float16*)carve((size_t)BATCH * HID * NODE * 2);
    float*    T2T      = (float*)carve((size_t)BATCH * HID * NODE * 4);

    const dim3 blk256(256);
    const int g_pack = (BATCH * HID * NODE) / 256;          // 32768
    const int g_wcat = (HID * KCAT) / 256;                  // 7168
    const int g_mx   = (BATCH * 7 * 4 * 32) / 8;            // 28672 (8 waves/block)
    const int g_big  = BATCH * 4;                           // 1024 (64x128 tile/block)

    // weights + adjacency prep
    build_wcat_kernel<<<g_wcat, blk256, 0, stream>>>(basis0, self0, W0T);
    build_wcat_kernel<<<g_wcat, blk256, 0, stream>>>(basis1, self1, W1T);
    build_m1_kernel<<<BATCH, blk256, 0, stream>>>(adj, comb0, comb1, M1);
    build_m2_kernel<<<BATCH, blk256, 0, stream>>>(adj, comb0, comb1, M2);

    // initial node features + first graph input
    initial_kernel<<<dim3(BATCH, 2), blk256, 0, stream>>>(
        player, Ax, Ay, Bx, By, emb, coord_W, coord_b, in_W, in_b, initialH);
    pack_x1_kernel<<<g_pack, blk256, 0, stream>>>(enc, initialH, XT1);

    // ---- rgcn call #1 ----
    gemm_mx_kernel<<<g_mx, blk256, 0, stream>>>(M1, 0, XT1, Ycat);
    gemm_big_kernel<<<g_big, blk256, 0, stream>>>(Ycat, W0T, HT, nullptr, /*relu*/1);
    gemm_mx_kernel<<<g_mx, blk256, 0, stream>>>(M1, 1, HT, Ycat);
    gemm_big_kernel<<<g_big, blk256, 0, stream>>>(Ycat, W1T, T1T, nullptr, /*sigmoid*/2);

    // ---- rgcn call #2 (63 nodes: passed rows 0..61 + initial[:,1]) ----
    pack_x2_kernel<<<g_pack, blk256, 0, stream>>>(enc, T1T, initialH, XT2);
    gemm_mx_kernel<<<g_mx, blk256, 0, stream>>>(M2, 0, XT2, Ycat);
    gemm_big_kernel<<<g_big, blk256, 0, stream>>>(Ycat, W0T, HT, nullptr, /*relu*/1);
    gemm_mx_kernel<<<g_mx, blk256, 0, stream>>>(M2, 1, HT, Ycat);
    gemm_big_kernel<<<g_big, blk256, 0, stream>>>(Ycat, W1T, nullptr, T2T, /*sigmoid*/2);

    // ---- logits ----
    final_kernel<<<BATCH, 32, 0, stream>>>(T2T, type_W, type_b, out);
}
